// RobustSlotAttention_18184891531792
// MI455X (gfx1250) — compile-verified, matrix operand order checked by reading, not run on hardware
//
#include <hip/hip_runtime.h>
#include <hip/hip_bf16.h>

typedef __attribute__((ext_vector_type(16))) _Float16 v16h;
typedef __attribute__((ext_vector_type(8)))  float    v8f;
typedef __attribute__((ext_vector_type(4)))  int      i4v;

#define Bn   64
#define Nn   1024
#define INd  512
#define Sn   12
#define Sp   16
#define Dn   256
#define Hn   512
#define SCALE_ATTN 0.0625f   /* 256^-0.5 */

#define AS1 __attribute__((address_space(1)))
#define AS3 __attribute__((address_space(3)))

#if __has_builtin(__builtin_amdgcn_global_load_async_to_lds_b128) && \
    __has_builtin(__builtin_amdgcn_s_wait_asynccnt)
#define HAVE_ASYNC_LDS 1
#else
#define HAVE_ASYNC_LDS 0
#endif

__device__ __forceinline__ float gelu_f(float x) {
    return 0.5f * x * (1.0f + erff(x * 0.70710678118654752440f));
}
__device__ __forceinline__ float sigmoid_f(float x) {
    return 1.0f / (1.0f + expf(-x));
}

// ---------------- weight prep ----------------
__global__ void k_trans_f16(const float* __restrict__ W, _Float16* __restrict__ Wt,
                            int K, int N) {
    long idx = (long)blockIdx.x * 256 + threadIdx.x;
    if (idx >= (long)K * N) return;
    long n = idx / K, kk = idx % K;
    Wt[idx] = (_Float16)W[kk * (long)N + n];
}
__global__ void k_conv_f16(const float* __restrict__ W, _Float16* __restrict__ Wt, long count) {
    long idx = (long)blockIdx.x * 256 + threadIdx.x;
    if (idx < count) Wt[idx] = (_Float16)W[idx];
}
__global__ void k_trans_f32(const float* __restrict__ W, float* __restrict__ Wt,
                            int K, int N) {
    long idx = (long)blockIdx.x * 256 + threadIdx.x;
    if (idx >= (long)K * N) return;
    long n = idx / K, kk = idx % K;
    Wt[idx] = W[kk * (long)N + n];
}

// =====================================================================
// WMMA GEMM, 1 N-tile per wave (used only for the S=16 logits GEMM)
// C[b][M][N] = alpha*(A[b][M][K] x Bt[b][N][K]^T) + bias[N]
// =====================================================================
__global__ void k_gemm1(const _Float16* __restrict__ A,
                        const _Float16* __restrict__ Bt,
                        const float* __restrict__ bias,
                        float* __restrict__ Cf, _Float16* __restrict__ Ch,
                        int M, int N, int K,
                        long sA, long sB, long sC, float alpha) {
    int wave = threadIdx.x >> 5;
    int lane = threadIdx.x & 31;
    int tm = blockIdx.x * 4 + wave;
    if (tm * 16 >= M) return;
    int tn = blockIdx.y;
    int b  = blockIdx.z;
    int r15 = lane & 15;
    int sub = lane >> 4;

    const _Float16* Ap = A + (long)b * sA + ((long)tm * 16 + r15) * (long)K;
    const _Float16* Bp = Bt + (long)b * sB + ((long)tn * 16 + r15) * (long)K;

    v8f acc = {};
    for (int k0 = 0; k0 < K; k0 += 32) {
        const _Float16* pa = Ap + k0 + sub * 8;
        const _Float16* pb = Bp + k0 + sub * 8;
        v16h av, bv;
        ((uint4*)&av)[0] = *(const uint4*)(pa);
        ((uint4*)&av)[1] = *(const uint4*)(pa + 16);
        ((uint4*)&bv)[0] = *(const uint4*)(pb);
        ((uint4*)&bv)[1] = *(const uint4*)(pb + 16);
        acc = __builtin_amdgcn_wmma_f32_16x16x32_f16(false, av, false, bv,
                                                     (short)0, acc, false, false);
    }

    int col = tn * 16 + r15;
    float bb = bias ? bias[col] : 0.0f;
#pragma unroll
    for (int r = 0; r < 8; ++r) {
        float ov = acc[r] * alpha + bb;
        long idx = (long)b * sC + ((long)tm * 16 + r + 8 * sub) * (long)N + col;
        if (Cf) Cf[idx] = ov;
        if (Ch) Ch[idx] = (_Float16)ov;
    }
}

// =====================================================================
// WMMA GEMM, 4 N-tiles per wave + LDS-staged B (async copy, double buffer)
// Block = 128 threads (4 waves along M).  Block tile = 64(M) x 64(N).
// Requires N % 64 == 0, K % 32 == 0, M % 16 == 0.
// outputs: Cf f32, Ch f16 row-major, Ct f16 transposed (Ct[b][col*ldct+row])
// =====================================================================
__global__ void k_gemm4(const _Float16* __restrict__ A,
                        const _Float16* __restrict__ Bt,
                        const float* __restrict__ bias,
                        float* __restrict__ Cf, _Float16* __restrict__ Ch,
                        _Float16* __restrict__ Ct,
                        int M, int N, int K,
                        long sA, long sB, long sC, long sCt, int ldct,
                        float alpha) {
    __shared__ _Float16 ldsB[2][64 * 40];      // 64 rows, padded stride 40 halves

    int wave = threadIdx.x >> 5;
    int lane = threadIdx.x & 31;
    int tm = blockIdx.x * 4 + wave;
    int b  = blockIdx.z;
    int r15 = lane & 15;
    int sub = lane >> 4;
    bool active = (tm * 16 < M);

    const _Float16* Ap = A + (long)b * sA +
                         ((long)(active ? tm : 0) * 16 + r15) * (long)K;
    long colBase = (long)blockIdx.y * 64;

    // cooperative stage of one 64x32 B chunk into ldsB[buf]
    auto stage = [&](int buf, int k0) {
#pragma unroll
        for (int t = 0; t < 2; ++t) {
            int id = threadIdx.x * 2 + t;          // 0..255
            int row = id >> 2, seg = id & 3;       // 4 x 16B per row
            const _Float16* src = Bt + (long)b * sB + (colBase + row) * (long)K
                                  + k0 + seg * 8;
            _Float16* dst = &ldsB[buf][row * 40 + seg * 8];
#if HAVE_ASYNC_LDS
            __builtin_amdgcn_global_load_async_to_lds_b128(
                (AS1 i4v*)src, (AS3 i4v*)dst, 0, 0);
#else
            *(uint4*)dst = *(const uint4*)src;
#endif
        }
    };

    v8f acc0 = {}, acc1 = {}, acc2 = {}, acc3 = {};
    int nk = K >> 5;

    stage(0, 0);
    for (int i = 0; i < nk; ++i) {
        int cur = i & 1;
        bool more = (i + 1 < nk);
        if (more) stage(cur ^ 1, (i + 1) << 5);
#if HAVE_ASYNC_LDS
        if (more) __builtin_amdgcn_s_wait_asynccnt(2);
        else      __builtin_amdgcn_s_wait_asynccnt(0);
#endif
        __syncthreads();
        if (active) {
            int k0 = i << 5;
            const _Float16* pa = Ap + k0 + sub * 8;
            v16h av;
            ((uint4*)&av)[0] = *(const uint4*)(pa);
            ((uint4*)&av)[1] = *(const uint4*)(pa + 16);
#pragma unroll
            for (int t = 0; t < 4; ++t) {
                const _Float16* pb = &ldsB[cur][(t * 16 + r15) * 40 + sub * 8];
                v16h bv;
                ((uint4*)&bv)[0] = *(const uint4*)(pb);
                ((uint4*)&bv)[1] = *(const uint4*)(pb + 16);
                if (t == 0)
                    acc0 = __builtin_amdgcn_wmma_f32_16x16x32_f16(false, av, false, bv,
                                                                  (short)0, acc0, false, false);
                else if (t == 1)
                    acc1 = __builtin_amdgcn_wmma_f32_16x16x32_f16(false, av, false, bv,
                                                                  (short)0, acc1, false, false);
                else if (t == 2)
                    acc2 = __builtin_amdgcn_wmma_f32_16x16x32_f16(false, av, false, bv,
                                                                  (short)0, acc2, false, false);
                else
                    acc3 = __builtin_amdgcn_wmma_f32_16x16x32_f16(false, av, false, bv,
                                                                  (short)0, acc3, false, false);
            }
        }
        __syncthreads();
    }

    if (!active) return;

#pragma unroll
    for (int t = 0; t < 4; ++t) {
        v8f acc = (t == 0) ? acc0 : (t == 1) ? acc1 : (t == 2) ? acc2 : acc3;
        int col = (int)colBase + t * 16 + r15;
        float bb = bias ? bias[col] : 0.0f;
        float outv[8];
#pragma unroll
        for (int r = 0; r < 8; ++r) outv[r] = acc[r] * alpha + bb;
        if (Cf) {
            float* cp = Cf + (long)b * sC + ((long)tm * 16) * (long)N + col;
#pragma unroll
            for (int r = 0; r < 8; ++r) cp[(long)(r + 8 * sub) * N] = outv[r];
        }
        if (Ch) {
            _Float16* cp = Ch + (long)b * sC + ((long)tm * 16) * (long)N + col;
#pragma unroll
            for (int r = 0; r < 8; ++r) cp[(long)(r + 8 * sub) * N] = (_Float16)outv[r];
        }
        if (Ct) {
            alignas(16) _Float16 t8[8];
#pragma unroll
            for (int r = 0; r < 8; ++r) t8[r] = (_Float16)outv[r];
            long off = (long)b * sCt + (long)col * ldct + (long)tm * 16 + 8 * sub;
            *(uint4*)(Ct + off) = *(const uint4*)t8;
        }
    }
}

// ---------------- LayerNorm (rows x L), f16 out, optional GELU ----------------
__global__ void k_ln(const void* __restrict__ X, int xIsF16,
                     const float* __restrict__ g, const float* __restrict__ bt,
                     _Float16* __restrict__ o16, int L, int per, int doGelu) {
    __shared__ float red[256];
    int row = blockIdx.x, t = threadIdx.x;
    long base = (long)row * L;
    float v[4];
    float s = 0.f;
    for (int i = 0; i < per; ++i) {
        int idx = t + i * 256;
        v[i] = xIsF16 ? (float)((const _Float16*)X)[base + idx]
                      : ((const float*)X)[base + idx];
        s += v[i];
    }
    red[t] = s; __syncthreads();
    for (int o = 128; o > 0; o >>= 1) { if (t < o) red[t] += red[t + o]; __syncthreads(); }
    float mean = red[0] / (float)L;
    __syncthreads();
    float s2 = 0.f;
    for (int i = 0; i < per; ++i) { float d = v[i] - mean; s2 += d * d; }
    red[t] = s2; __syncthreads();
    for (int o = 128; o > 0; o >>= 1) { if (t < o) red[t] += red[t + o]; __syncthreads(); }
    float inv = rsqrtf(red[0] / (float)L + 1e-5f);
    for (int i = 0; i < per; ++i) {
        int idx = t + i * 256;
        float y = (v[i] - mean) * inv * g[idx] + bt[idx];
        if (doGelu) y = gelu_f(y);
        o16[base + idx] = (_Float16)y;
    }
}

// ---------------- slot init ----------------
__global__ void k_init_slots(const float* __restrict__ mu, const float* __restrict__ sigma,
                             const float* __restrict__ noise,
                             float* __restrict__ slots, _Float16* __restrict__ s16) {
    int idx = blockIdx.x * 256 + threadIdx.x;           // B*Sp*D = 262144
    if (idx >= Bn * Sp * Dn) return;
    int b = idx >> 12, rem = idx & 4095, s = rem >> 8, d = rem & 255;
    float val = 0.f;
    if (s < Sn) val = mu[s * Dn + d] + sigma[s * Dn + d] * noise[(long)b * Sn * Dn + s * Dn + d];
    slots[idx] = val; s16[idx] = (_Float16)val;
}

// ---------------- softmax over S (writes transposed f16 attn) ----------------
__global__ void k_softmax(const float* __restrict__ logits, _Float16* __restrict__ attnT) {
    int idx = blockIdx.x * 256 + threadIdx.x;           // B*N = 65536
    if (idx >= Bn * Nn) return;
    int b = idx >> 10, n = idx & 1023;
    const float* lp = logits + (long)idx * Sp;
    float mx = -1e30f;
    for (int s = 0; s < Sn; ++s) mx = fmaxf(mx, lp[s]);
    float e[Sn]; float sum = 0.f;
    for (int s = 0; s < Sn; ++s) { e[s] = expf(lp[s] - mx); sum += e[s]; }
    float inv = 1.0f / sum;
    _Float16* ap = attnT + (long)b * (Sp * Nn) + n;
    for (int s = 0; s < Sn; ++s) ap[(long)s * Nn] = (_Float16)(e[s] * inv + 1e-8f);
    for (int s = Sn; s < Sp; ++s) ap[(long)s * Nn] = (_Float16)0.0f;
}

__global__ void k_colsum(const _Float16* __restrict__ attnT, float* __restrict__ colsum) {
    int idx = blockIdx.x * 256 + threadIdx.x;           // B*Sp = 1024
    if (idx >= Bn * Sp) return;
    const _Float16* ap = attnT + (long)idx * Nn;
    float s = 0.f;
    for (int n = 0; n < Nn; ++n) s += (float)ap[n];
    colsum[idx] = s;
}

__global__ void k_renorm(const float* __restrict__ raw, const float* __restrict__ colsum,
                         _Float16* __restrict__ o16) {
    int idx = blockIdx.x * 256 + threadIdx.x;           // B*Sp*D
    if (idx >= Bn * Sp * Dn) return;
    int bs = idx >> 8;
    float cs = colsum[bs];
    float v = (cs > 0.f) ? raw[idx] / cs : 0.f;
    o16[idx] = (_Float16)v;
}

// ---------------- GRU combine ----------------
__global__ void k_gru(const float* __restrict__ gi, const float* __restrict__ gh,
                      float* __restrict__ slots, _Float16* __restrict__ s16) {
    int idx = blockIdx.x * 256 + threadIdx.x;           // B*Sp*D
    if (idx >= Bn * Sp * Dn) return;
    int bs = idx >> 8, d = idx & 255;
    long gb = (long)bs * (3 * Dn);
    float r  = sigmoid_f(gi[gb + d]          + gh[gb + d]);
    float z  = sigmoid_f(gi[gb + Dn + d]     + gh[gb + Dn + d]);
    float nn = tanhf    (gi[gb + 2 * Dn + d] + r * gh[gb + 2 * Dn + d]);
    float sp = slots[idx];
    float sn = (1.0f - z) * nn + z * sp;
    slots[idx] = sn; s16[idx] = (_Float16)sn;
}

__global__ void k_gelu(const float* __restrict__ x, _Float16* __restrict__ o, long count) {
    long idx = (long)blockIdx.x * 256 + threadIdx.x;
    if (idx < count) o[idx] = (_Float16)gelu_f(x[idx]);
}

__global__ void k_add_slots(const float* __restrict__ mlp, float* __restrict__ slots,
                            _Float16* __restrict__ s16) {
    int idx = blockIdx.x * 256 + threadIdx.x;
    if (idx >= Bn * Sp * Dn) return;
    float v = slots[idx] + mlp[idx];
    slots[idx] = v; s16[idx] = (_Float16)v;
}

// ---------------- decoder / confidence helpers ----------------
__global__ void k_gather_sf(const float* __restrict__ slots, _Float16* __restrict__ sf) {
    int idx = blockIdx.x * 256 + threadIdx.x;           // B*S*D = 196608
    if (idx >= Bn * Sn * Dn) return;
    int b = idx / (Sn * Dn), k = idx % (Sn * Dn);
    sf[idx] = (_Float16)slots[(long)b * (Sp * Dn) + k];
}

__global__ void k_target(const float* __restrict__ inputs, float* __restrict__ target) {
    int idx = blockIdx.x * 256 + threadIdx.x;           // B*IN = 32768
    if (idx >= Bn * INd) return;
    int b = idx >> 9, d = idx & 511;
    const float* p = inputs + (long)b * Nn * INd + d;
    float s = 0.f;
    for (int n = 0; n < Nn; ++n) s += p[(long)n * INd];
    target[idx] = s * (1.0f / (float)Nn);
}

__global__ void k_recon_err(const float* __restrict__ recon, const float* __restrict__ target,
                            float* __restrict__ err) {
    __shared__ float red[256];
    int b = blockIdx.x, t = threadIdx.x;
    float d0 = recon[(long)b * INd + t]       - target[(long)b * INd + t];
    float d1 = recon[(long)b * INd + t + 256] - target[(long)b * INd + t + 256];
    red[t] = d0 * d0 + d1 * d1; __syncthreads();
    for (int o = 128; o > 0; o >>= 1) { if (t < o) red[t] += red[t + o]; __syncthreads(); }
    if (t == 0) err[b] = red[0] / (float)INd;
}

__global__ void k_conf1(const float* __restrict__ slots, const float* __restrict__ err,
                        const float* __restrict__ W1t, const float* __restrict__ b1,
                        float* __restrict__ ch) {
    int idx = blockIdx.x * 256 + threadIdx.x;           // B*H = 32768
    if (idx >= Bn * Hn) return;
    int b = idx >> 9, h = idx & 511;
    const float* wp = W1t + (long)h * (Sn * Dn + 1);
    const float* sp = slots + (long)b * (Sp * Dn);
    float acc = b1[h];
    for (int k = 0; k < Sn * Dn; ++k) acc += sp[k] * wp[k];
    acc += err[b] * wp[Sn * Dn];
    ch[idx] = gelu_f(acc);
}

__global__ void k_out_slots(const float* __restrict__ slots, float* __restrict__ out) {
    int idx = blockIdx.x * 256 + threadIdx.x;           // 196608
    if (idx >= Bn * Sn * Dn) return;
    int b = idx / (Sn * Dn), k = idx % (Sn * Dn);
    out[idx] = slots[(long)b * (Sp * Dn) + k];
}

__global__ void k_conf2(const float* __restrict__ ch, const float* __restrict__ W2,
                        const float* __restrict__ b2, float* __restrict__ out) {
    __shared__ float red[256];
    int b = blockIdx.x, t = threadIdx.x;
    red[t] = ch[(long)b * Hn + t] * W2[t] + ch[(long)b * Hn + t + 256] * W2[t + 256];
    __syncthreads();
    for (int o = 128; o > 0; o >>= 1) { if (t < o) red[t] += red[t + o]; __syncthreads(); }
    if (t == 0) out[(long)Bn * Sn * Dn + b] = sigmoid_f(red[0] + b2[0]);
}

// ---------------- host side ----------------
static inline void gemm4(hipStream_t st, const _Float16* A, const _Float16* Bt,
                         const float* bias, float* Cf, _Float16* Ch, _Float16* Ct,
                         int M, int N, int K, int batch,
                         long sA, long sB, long sC, long sCt, int ldct, float alpha) {
    dim3 g((M + 63) / 64, N / 64, batch), b(128);
    k_gemm4<<<g, b, 0, st>>>(A, Bt, bias, Cf, Ch, Ct, M, N, K,
                             sA, sB, sC, sCt, ldct, alpha);
}
static inline void gemm1(hipStream_t st, const _Float16* A, const _Float16* Bt,
                         const float* bias, float* Cf, _Float16* Ch,
                         int M, int N, int K, int batch,
                         long sA, long sB, long sC, float alpha) {
    dim3 g((M + 63) / 64, N / 16, batch), b(128);
    k_gemm1<<<g, b, 0, st>>>(A, Bt, bias, Cf, Ch, M, N, K, sA, sB, sC, alpha);
}
#define ELG(n) dim3((unsigned)(((n) + 255) / 256)), dim3(256)

extern "C" void kernel_launch(void* const* d_in, const int* in_sizes, int n_in,
                              void* d_out, int out_size, void* d_ws, size_t ws_size,
                              hipStream_t stream) {
    (void)in_sizes; (void)n_in; (void)out_size; (void)ws_size;
    const float* inputs   = (const float*)d_in[0];
    const float* noise    = (const float*)d_in[1];
    const float* slots_mu = (const float*)d_in[2];
    const float* slots_sg = (const float*)d_in[3];
    const float* W_in = (const float*)d_in[4];  const float* b_in = (const float*)d_in[5];
    const float* Wk = (const float*)d_in[6];    const float* bk = (const float*)d_in[7];
    const float* Wq = (const float*)d_in[8];    const float* bq = (const float*)d_in[9];
    const float* Wv = (const float*)d_in[10];   const float* bv = (const float*)d_in[11];
    const float* gWih = (const float*)d_in[12]; const float* gWhh = (const float*)d_in[13];
    const float* gbih = (const float*)d_in[14]; const float* gbhh = (const float*)d_in[15];
    const float* mW1 = (const float*)d_in[16];  const float* mb1 = (const float*)d_in[17];
    const float* mW2 = (const float*)d_in[18];  const float* mb2 = (const float*)d_in[19];
    const float* ln_in_g = (const float*)d_in[20];   const float* ln_in_b = (const float*)d_in[21];
    const float* ln_sl_g = (const float*)d_in[22];   const float* ln_sl_b = (const float*)d_in[23];
    const float* ln_ml_g = (const float*)d_in[24];   const float* ln_ml_b = (const float*)d_in[25];
    const float* dW1 = (const float*)d_in[26];  const float* db1 = (const float*)d_in[27];
    const float* dlg = (const float*)d_in[28];  const float* dlb = (const float*)d_in[29];
    const float* dW2 = (const float*)d_in[30];  const float* db2 = (const float*)d_in[31];
    const float* dW3 = (const float*)d_in[32];  const float* db3 = (const float*)d_in[33];
    const float* cW1 = (const float*)d_in[34];  const float* cb1 = (const float*)d_in[35];
    const float* cW2 = (const float*)d_in[36];  const float* cb2 = (const float*)d_in[37];
    float* out = (float*)d_out;

    char* p = (char*)d_ws;
    auto take = [&](size_t bytes) -> void* {
        void* r = (void*)p; p += (bytes + 255) & ~(size_t)255; return r;
    };
    _Float16* h_Wt_in = (_Float16*)take((size_t)Dn * INd * 2);
    _Float16* h_Wkt   = (_Float16*)take((size_t)Dn * Dn * 2);
    _Float16* h_Wqt   = (_Float16*)take((size_t)Dn * Dn * 2);
    _Float16* h_Wvt   = (_Float16*)take((size_t)Dn * Dn * 2);
    _Float16* h_Wih   = (_Float16*)take((size_t)3 * Dn * Dn * 2);
    _Float16* h_Whh   = (_Float16*)take((size_t)3 * Dn * Dn * 2);
    _Float16* h_W1t   = (_Float16*)take((size_t)Hn * Dn * 2);
    _Float16* h_W2t   = (_Float16*)take((size_t)Dn * Hn * 2);
    _Float16* h_d1t   = (_Float16*)take((size_t)(2 * Hn) * (Sn * Dn) * 2);
    _Float16* h_d2t   = (_Float16*)take((size_t)Hn * (2 * Hn) * 2);
    _Float16* h_d3t   = (_Float16*)take((size_t)INd * Hn * 2);
    float*    f_c1t   = (float*)take((size_t)Hn * (Sn * Dn + 1) * 4);

    _Float16* h_in  = (_Float16*)take((size_t)Bn * Nn * INd * 2);
    _Float16* h_x   = (_Float16*)take((size_t)Bn * Nn * Dn * 2);
    _Float16* h_k   = (_Float16*)take((size_t)Bn * Nn * Dn * 2);
    _Float16* h_vt  = (_Float16*)take((size_t)Bn * Dn * Nn * 2);
    float*    f_sl  = (float*)take((size_t)Bn * Sp * Dn * 4);
    _Float16* h_sl  = (_Float16*)take((size_t)Bn * Sp * Dn * 2);
    _Float16* h_sn  = (_Float16*)take((size_t)Bn * Sp * Dn * 2);
    _Float16* h_q   = (_Float16*)take((size_t)Bn * Sp * Dn * 2);
    float*    f_lg  = (float*)take((size_t)Bn * Nn * Sp * 4);
    _Float16* h_at  = (_Float16*)take((size_t)Bn * Sp * Nn * 2);
    float*    f_cs  = (float*)take((size_t)Bn * Sp * 4);
    float*    f_ur  = (float*)take((size_t)Bn * Sp * Dn * 4);
    _Float16* h_up  = (_Float16*)take((size_t)Bn * Sp * Dn * 2);
    float*    f_gi  = (float*)take((size_t)Bn * Sp * 3 * Dn * 4);
    float*    f_gh  = (float*)take((size_t)Bn * Sp * 3 * Dn * 4);
    _Float16* h_m   = (_Float16*)take((size_t)Bn * Sp * Dn * 2);
    float*    f_h1  = (float*)take((size_t)Bn * Sp * Hn * 4);
    _Float16* h_h1  = (_Float16*)take((size_t)Bn * Sp * Hn * 2);
    float*    f_mo  = (float*)take((size_t)Bn * Sp * Dn * 4);
    _Float16* h_sf  = (_Float16*)take((size_t)Bn * Sn * Dn * 2);
    float*    f_dh  = (float*)take((size_t)Bn * 2 * Hn * 4);
    _Float16* h_dh  = (_Float16*)take((size_t)Bn * 2 * Hn * 2);
    float*    f_d2  = (float*)take((size_t)Bn * Hn * 4);
    _Float16* h_d2  = (_Float16*)take((size_t)Bn * Hn * 2);
    float*    f_rc  = (float*)take((size_t)Bn * INd * 4);
    float*    f_tg  = (float*)take((size_t)Bn * INd * 4);
    float*    f_er  = (float*)take((size_t)Bn * 4);
    float*    f_ch  = (float*)take((size_t)Bn * Hn * 4);

    // ---- weight prep (f16 transpose / convert) ----
    k_trans_f16<<<ELG(INd * Dn), 0, stream>>>(W_in, h_Wt_in, INd, Dn);
    k_trans_f16<<<ELG(Dn * Dn), 0, stream>>>(Wk, h_Wkt, Dn, Dn);
    k_trans_f16<<<ELG(Dn * Dn), 0, stream>>>(Wq, h_Wqt, Dn, Dn);
    k_trans_f16<<<ELG(Dn * Dn), 0, stream>>>(Wv, h_Wvt, Dn, Dn);
    k_conv_f16<<<ELG(3 * Dn * Dn), 0, stream>>>(gWih, h_Wih, (long)3 * Dn * Dn);
    k_conv_f16<<<ELG(3 * Dn * Dn), 0, stream>>>(gWhh, h_Whh, (long)3 * Dn * Dn);
    k_trans_f16<<<ELG(Dn * Hn), 0, stream>>>(mW1, h_W1t, Dn, Hn);
    k_trans_f16<<<ELG(Hn * Dn), 0, stream>>>(mW2, h_W2t, Hn, Dn);
    k_trans_f16<<<ELG(Sn * Dn * 2 * Hn), 0, stream>>>(dW1, h_d1t, Sn * Dn, 2 * Hn);
    k_trans_f16<<<ELG(2 * Hn * Hn), 0, stream>>>(dW2, h_d2t, 2 * Hn, Hn);
    k_trans_f16<<<ELG(Hn * INd), 0, stream>>>(dW3, h_d3t, Hn, INd);
    k_trans_f32<<<ELG((Sn * Dn + 1) * Hn), 0, stream>>>(cW1, f_c1t, Sn * Dn + 1, Hn);
    k_conv_f16<<<ELG(Bn * Nn * INd), 0, stream>>>(inputs, h_in, (long)Bn * Nn * INd);

    // ---- encoder: x = LN(inputs@W_in + b_in) ; k = x@Wk ; v^T = (x@Wv)^T ----
    gemm4(stream, h_in, h_Wt_in, b_in, nullptr, h_x, nullptr,
          Bn * Nn, Dn, INd, 1, 0, 0, 0, 0, 0, 1.0f);
    k_ln<<<Bn * Nn, 256, 0, stream>>>(h_x, 1, ln_in_g, ln_in_b, h_x, Dn, 1, 0);
    gemm4(stream, h_x, h_Wkt, bk, nullptr, h_k, nullptr,
          Bn * Nn, Dn, Dn, 1, 0, 0, 0, 0, 0, 1.0f);
    gemm4(stream, h_x, h_Wvt, bv, nullptr, nullptr, h_vt,
          Nn, Dn, Dn, Bn, (long)Nn * Dn, 0, 0, (long)Dn * Nn, Nn, 1.0f);

    k_init_slots<<<ELG(Bn * Sp * Dn), 0, stream>>>(slots_mu, slots_sg, noise, f_sl, h_sl);

    for (int it = 0; it < 3; ++it) {
        k_ln<<<Bn * Sp, 256, 0, stream>>>(f_sl, 0, ln_sl_g, ln_sl_b, h_sn, Dn, 1, 0);
        gemm4(stream, h_sn, h_Wqt, bq, nullptr, h_q, nullptr,
              Bn * Sp, Dn, Dn, 1, 0, 0, 0, 0, 0, 1.0f);
        gemm1(stream, h_k, h_q, nullptr, f_lg, nullptr,
              Nn, Sp, Dn, Bn, (long)Nn * Dn, (long)Sp * Dn, (long)Nn * Sp, SCALE_ATTN);
        k_softmax<<<ELG(Bn * Nn), 0, stream>>>(f_lg, h_at);
        k_colsum<<<ELG(Bn * Sp), 0, stream>>>(h_at, f_cs);
        gemm4(stream, h_at, h_vt, nullptr, f_ur, nullptr, nullptr,
              Sp, Dn, Nn, Bn, (long)Sp * Nn, (long)Dn * Nn, (long)Sp * Dn, 0, 0, 1.0f);
        k_renorm<<<ELG(Bn * Sp * Dn), 0, stream>>>(f_ur, f_cs, h_up);
        gemm4(stream, h_up, h_Wih, gbih, f_gi, nullptr, nullptr,
              Bn * Sp, 3 * Dn, Dn, 1, 0, 0, 0, 0, 0, 1.0f);
        gemm4(stream, h_sl, h_Whh, gbhh, f_gh, nullptr, nullptr,
              Bn * Sp, 3 * Dn, Dn, 1, 0, 0, 0, 0, 0, 1.0f);
        k_gru<<<ELG(Bn * Sp * Dn), 0, stream>>>(f_gi, f_gh, f_sl, h_sl);
        k_ln<<<Bn * Sp, 256, 0, stream>>>(f_sl, 0, ln_ml_g, ln_ml_b, h_m, Dn, 1, 0);
        gemm4(stream, h_m, h_W1t, mb1, f_h1, nullptr, nullptr,
              Bn * Sp, Hn, Dn, 1, 0, 0, 0, 0, 0, 1.0f);
        k_gelu<<<ELG(Bn * Sp * Hn), 0, stream>>>(f_h1, h_h1, (long)Bn * Sp * Hn);
        gemm4(stream, h_h1, h_W2t, mb2, f_mo, nullptr, nullptr,
              Bn * Sp, Dn, Hn, 1, 0, 0, 0, 0, 0, 1.0f);
        k_add_slots<<<ELG(Bn * Sp * Dn), 0, stream>>>(f_mo, f_sl, h_sl);
    }

    // ---- decoder ----
    k_gather_sf<<<ELG(Bn * Sn * Dn), 0, stream>>>(f_sl, h_sf);
    gemm4(stream, h_sf, h_d1t, db1, f_dh, nullptr, nullptr,
          Bn, 2 * Hn, Sn * Dn, 1, 0, 0, 0, 0, 0, 1.0f);
    k_ln<<<Bn, 256, 0, stream>>>(f_dh, 0, dlg, dlb, h_dh, 2 * Hn, 4, 1);
    gemm4(stream, h_dh, h_d2t, db2, f_d2, nullptr, nullptr,
          Bn, Hn, 2 * Hn, 1, 0, 0, 0, 0, 0, 1.0f);
    k_gelu<<<ELG(Bn * Hn), 0, stream>>>(f_d2, h_d2, (long)Bn * Hn);
    gemm4(stream, h_d2, h_d3t, db3, f_rc, nullptr, nullptr,
          Bn, INd, Hn, 1, 0, 0, 0, 0, 0, 1.0f);

    // ---- confidence + outputs ----
    k_target<<<ELG(Bn * INd), 0, stream>>>(inputs, f_tg);
    k_recon_err<<<Bn, 256, 0, stream>>>(f_rc, f_tg, f_er);
    k_conf1<<<ELG(Bn * Hn), 0, stream>>>(f_sl, f_er, f_c1t, cb1, f_ch);
    k_out_slots<<<ELG(Bn * Sn * Dn), 0, stream>>>(f_sl, out);
    k_conf2<<<Bn, 256, 0, stream>>>(f_ch, cW2, cb2, out);
}
